// MultiHeadMGN_1632087573298
// MI455X (gfx1250) — matrix operand order repaired and checked by reading, not compiled, not run on hardware
//
#include <hip/hip_runtime.h>
#include <hip/hip_bf16.h>
#include <math.h>

// ---------------------------------------------------------------------------
// CDNA5 (gfx1250) MeshGraphNet forward.
// All 128-wide MLPs run on v_wmma_f32_16x16x32_bf16 (wave32), f32 accumulate.
// segment_sum via global_atomic_add_f32 into L2-resident agg buffer.
// ---------------------------------------------------------------------------

typedef __attribute__((ext_vector_type(16))) __bf16 bf16x16;
typedef __attribute__((ext_vector_type(8)))  float  floatx8;

union FragB16 { bf16x16 v; unsigned int u[8]; };

__device__ __forceinline__ unsigned short f2bf(float f) {
  union { float f; unsigned int u; } c; c.f = f;
  unsigned int r = c.u + 0x7FFFu + ((c.u >> 16) & 1u);  // round-to-nearest-even
  return (unsigned short)(r >> 16);
}
__device__ __forceinline__ float bf2f(unsigned short b) {
  union { unsigned int u; float f; } c; c.u = ((unsigned int)b) << 16;
  return c.f;
}
__device__ __forceinline__ float silu_f(float x) { return x / (1.0f + expf(-x)); }

__device__ __forceinline__ floatx8 wmma_bf16(FragB16 a, FragB16 b, floatx8 c) {
  // D = A(16x32 bf16) * B(32x16 bf16) + C(16x16 f32)
  return __builtin_amdgcn_wmma_f32_16x16x32_bf16(false, a.v, false, b.v,
                                                 (short)0, c, false, false);
}

// B fragment, prepacked in global memory: frag = 32 lanes x 8 dwords.
__device__ __forceinline__ FragB16 load_b_frag(const unsigned int* __restrict__ packed,
                                               int frag, int lane) {
  FragB16 F;
  const uint4* p = (const uint4*)(packed + ((size_t)frag << 8) + (lane << 3));
  uint4 a = p[0], b = p[1];
  F.u[0] = a.x; F.u[1] = a.y; F.u[2] = a.z; F.u[3] = a.w;
  F.u[4] = b.x; F.u[5] = b.y; F.u[6] = b.z; F.u[7] = b.w;
  return F;
}

// A fragment from LDS activation tile (ushort bf16, [16][ld] row-major).
// 16-bit A 16x32 layout: lanes 0-15 hold K 0..7,16..23; lanes 16-31 hold K 8..15,24..31.
__device__ __forceinline__ FragB16 load_a_frag(const unsigned short* __restrict__ t,
                                               int ld, int k0, int lane) {
  FragB16 F;
  int m  = lane & 15;
  int kh = (lane >> 4) & 1;
  const unsigned short* row = t + m * ld + k0 + kh * 8;
#pragma unroll
  for (int v = 0; v < 8; ++v) {
    int k = ((v >> 2) << 4) + ((v & 3) << 1);   // (v/4)*16 + (v%4)*2
    F.u[v] = *(const unsigned int*)(row + k);   // ds_load_b32: pair (k, k+1)
  }
  return F;
}

// ---------------------------------------------------------------------------
// Weight pre-pack: f32 W[Ksrc][128] -> bf16 B fragments (Kpad/32)*8 frags.
// dst[f*256 + lane*8 + v] = pack(W[k][n], W[k+1][n]) per the 32x16 B layout.
// ---------------------------------------------------------------------------
__global__ void pack_w_kernel(const float* __restrict__ W, unsigned int* __restrict__ dst,
                              int Ksrc, int Kpad) {
  int tot = (Kpad / 32) * 8 * 256;
  for (int t = blockIdx.x * blockDim.x + threadIdx.x; t < tot;
       t += gridDim.x * blockDim.x) {
    int f = t >> 8, r = t & 255;
    int lane = r >> 3, v = r & 7;
    int kc = f >> 3, nt = f & 7;
    int k = kc * 32 + ((lane >> 4) & 1) * 8 + ((v >> 2) << 4) + ((v & 3) << 1);
    int n = nt * 16 + (lane & 15);
    float w0 = (k < Ksrc)     ? W[(size_t)k * 128 + n]       : 0.0f;
    float w1 = (k + 1 < Ksrc) ? W[(size_t)(k + 1) * 128 + n] : 0.0f;
    dst[t] = (unsigned int)f2bf(w0) | ((unsigned int)f2bf(w1) << 16);
  }
}

__global__ void zero_kernel(float* __restrict__ p, long n) {
  for (long i = blockIdx.x * (long)blockDim.x + threadIdx.x; i < n;
       i += (long)gridDim.x * blockDim.x)
    p[i] = 0.0f;
}

// ---------------------------------------------------------------------------
// Encoder: out[R][128] = MLP2(in[R][Kin]) with Kin in {16,4} padded to 32.
// 8 waves/block, one 16-row tile per wave.
// ---------------------------------------------------------------------------
__global__ void __launch_bounds__(256)
encoder_kernel(const float* __restrict__ in, int Kin, int R,
               const unsigned int* __restrict__ pw1, const float* __restrict__ b1,
               const unsigned int* __restrict__ pw2, const float* __restrict__ b2,
               unsigned short* __restrict__ out) {
  __shared__ unsigned short sA[8][16 * 32];
  __shared__ unsigned short sH[8][16 * 128];
  int lane = threadIdx.x & 31;
  int wave = threadIdx.x >> 5;
  long base = ((long)blockIdx.x * 8 + wave) * 16;
  unsigned short* A1 = sA[wave];
  unsigned short* H  = sH[wave];

  for (int idx = lane; idx < 16 * 32; idx += 32) {
    int r = idx >> 5, c = idx & 31;
    long row = base + r;
    float v = (c < Kin && row < R) ? in[row * Kin + c] : 0.0f;
    A1[idx] = f2bf(v);
  }
  __syncthreads();

  int n  = lane & 15;
  int mb = (lane >> 4) << 3;

  FragB16 a1 = load_a_frag(A1, 32, 0, lane);
#pragma unroll
  for (int nt = 0; nt < 8; ++nt) {
    floatx8 acc;
    float bias = b1[nt * 16 + n];
#pragma unroll
    for (int v = 0; v < 8; ++v) acc[v] = bias;
    FragB16 b = load_b_frag(pw1, nt, lane);
    acc = wmma_bf16(a1, b, acc);
#pragma unroll
    for (int v = 0; v < 8; ++v)
      H[(mb + v) * 128 + nt * 16 + n] = f2bf(silu_f(acc[v]));
  }
  __syncthreads();

#pragma unroll
  for (int nt = 0; nt < 8; ++nt) {
    floatx8 acc;
    float bias = b2[nt * 16 + n];
#pragma unroll
    for (int v = 0; v < 8; ++v) acc[v] = bias;
#pragma unroll
    for (int kc = 0; kc < 4; ++kc) {
      FragB16 a = load_a_frag(H, 128, kc * 32, lane);
      FragB16 b = load_b_frag(pw2, kc * 8 + nt, lane);
      acc = wmma_bf16(a, b, acc);
    }
#pragma unroll
    for (int v = 0; v < 8; ++v) {
      long row = base + mb + v;
      if (row < R) out[row * 128 + nt * 16 + n] = f2bf(acc[v]);
    }
  }
}

// ---------------------------------------------------------------------------
// Edge update (one layer): new_e = LN(MLP2([h[src]|h[dst]|e]) + e), scatter agg.
// 4 waves/block, one 16-edge tile per wave. Dynamic LDS.
// ---------------------------------------------------------------------------
__global__ void __launch_bounds__(128)
edge_update_kernel(const unsigned short* __restrict__ h,
                   unsigned short* __restrict__ e,
                   const int* __restrict__ srcI, const int* __restrict__ dstI,
                   const unsigned int* __restrict__ pw1, const float* __restrict__ b1,
                   const unsigned int* __restrict__ pw2, const float* __restrict__ b2,
                   const float* __restrict__ gam, const float* __restrict__ bet,
                   float* __restrict__ agg, int E) {
  extern __shared__ char smem[];
  const int CAT = 16 * 384, HID = 16 * 128, OUT = 16 * 128;
  unsigned short* catB = (unsigned short*)smem;
  unsigned short* hidB = catB + 4 * CAT;
  float* outB  = (float*)(hidB + 4 * HID);
  float* statB = outB + 4 * OUT;
  int*   dmB   = (int*)(statB + 4 * 32);

  int lane = threadIdx.x & 31;
  int wave = threadIdx.x >> 5;
  unsigned short* cat = catB + wave * CAT;
  unsigned short* hid = hidB + wave * HID;
  float* outf = outB + wave * OUT;
  float* stat = statB + wave * 32;
  int*   dm   = dmB + wave * 16;

  long base = ((long)blockIdx.x * 4 + wave) * 16;

  // stage [h[src] | h[dst] | e] as bf16, 8-byte gathers
  for (int i = 0; i < 16; ++i) {
    long eid = base + i;
    bool valid = eid < E;
    int s = valid ? srcI[eid] : 0;
    int d = valid ? dstI[eid] : 0;
    if (lane == 0) dm[i] = d;
    const unsigned long long* hs = (const unsigned long long*)(h + (size_t)s * 128);
    const unsigned long long* hd = (const unsigned long long*)(h + (size_t)d * 128);
    const unsigned long long* ee = (const unsigned long long*)(e + (size_t)eid * 128);
    unsigned long long* crow = (unsigned long long*)(cat + i * 384);
    crow[lane]      = valid ? hs[lane] : 0ull;
    crow[32 + lane] = valid ? hd[lane] : 0ull;
    crow[64 + lane] = valid ? ee[lane] : 0ull;
  }
  __syncthreads();

  int n  = lane & 15;
  int mb = (lane >> 4) << 3;

  // MLP layer 1: 384 -> 128, SiLU
#pragma unroll
  for (int nt = 0; nt < 8; ++nt) {
    floatx8 acc;
    float bias = b1[nt * 16 + n];
#pragma unroll
    for (int v = 0; v < 8; ++v) acc[v] = bias;
#pragma unroll
    for (int kc = 0; kc < 12; ++kc) {
      FragB16 a = load_a_frag(cat, 384, kc * 32, lane);
      FragB16 b = load_b_frag(pw1, kc * 8 + nt, lane);
      acc = wmma_bf16(a, b, acc);
    }
#pragma unroll
    for (int v = 0; v < 8; ++v)
      hid[(mb + v) * 128 + nt * 16 + n] = f2bf(silu_f(acc[v]));
  }
  __syncthreads();

  // MLP layer 2: 128 -> 128, + residual e
#pragma unroll
  for (int nt = 0; nt < 8; ++nt) {
    floatx8 acc;
    float bias = b2[nt * 16 + n];
#pragma unroll
    for (int v = 0; v < 8; ++v) acc[v] = bias;
#pragma unroll
    for (int kc = 0; kc < 4; ++kc) {
      FragB16 a = load_a_frag(hid, 128, kc * 32, lane);
      FragB16 b = load_b_frag(pw2, kc * 8 + nt, lane);
      acc = wmma_bf16(a, b, acc);
    }
#pragma unroll
    for (int v = 0; v < 8; ++v) {
      int m = mb + v;
      outf[m * 128 + nt * 16 + n] =
          acc[v] + bf2f(cat[m * 384 + 256 + nt * 16 + n]);
    }
  }
  __syncthreads();

  // LayerNorm stats (one row per lane 0..15)
  if (lane < 16) {
    const float* row = outf + lane * 128;
    float s = 0.0f, ss = 0.0f;
    for (int c = 0; c < 128; ++c) { float x = row[c]; s += x; ss += x * x; }
    float mean = s * (1.0f / 128.0f);
    float var  = ss * (1.0f / 128.0f) - mean * mean;
    stat[lane * 2]     = mean;
    stat[lane * 2 + 1] = rsqrtf(var + 1e-5f);
  }
  __syncthreads();

  // normalize, write new_e (bf16), scatter-add into agg (f32 atomics)
  for (int idx = lane; idx < 16 * 128; idx += 32) {
    int r = idx >> 7, c = idx & 127;
    long eid = base + r;
    if (eid < E) {
      float y = (outf[idx] - stat[r * 2]) * stat[r * 2 + 1] * gam[c] + bet[c];
      e[(size_t)eid * 128 + c] = f2bf(y);
      atomicAdd(agg + (size_t)dm[r] * 128 + c, y);
    }
  }
}

// ---------------------------------------------------------------------------
// Node update (one layer): h = LN(MLP2([h|agg]) + h), in place.
// ---------------------------------------------------------------------------
__global__ void __launch_bounds__(128)
node_update_kernel(unsigned short* __restrict__ h,
                   const float* __restrict__ agg,
                   const unsigned int* __restrict__ pw1, const float* __restrict__ b1,
                   const unsigned int* __restrict__ pw2, const float* __restrict__ b2,
                   const float* __restrict__ gam, const float* __restrict__ bet,
                   int N) {
  extern __shared__ char smem[];
  const int CAT = 16 * 256, HID = 16 * 128, OUT = 16 * 128;
  unsigned short* catB = (unsigned short*)smem;
  unsigned short* hidB = catB + 4 * CAT;
  float* outB  = (float*)(hidB + 4 * HID);
  float* statB = outB + 4 * OUT;

  int lane = threadIdx.x & 31;
  int wave = threadIdx.x >> 5;
  unsigned short* cat = catB + wave * CAT;
  unsigned short* hid = hidB + wave * HID;
  float* outf = outB + wave * OUT;
  float* stat = statB + wave * 32;

  long base = ((long)blockIdx.x * 4 + wave) * 16;

  for (int i = 0; i < 16; ++i) {
    long nid = base + i;
    bool valid = nid < N;
    const unsigned long long* hr = (const unsigned long long*)(h + (size_t)nid * 128);
    unsigned long long* crow = (unsigned long long*)(cat + i * 256);
    crow[lane] = valid ? hr[lane] : 0ull;
    const float4* ar = (const float4*)(agg + (size_t)nid * 128);
    float4 a4 = valid ? ar[lane] : make_float4(0.f, 0.f, 0.f, 0.f);
    unsigned long long pk = (unsigned long long)f2bf(a4.x)
                          | ((unsigned long long)f2bf(a4.y) << 16)
                          | ((unsigned long long)f2bf(a4.z) << 32)
                          | ((unsigned long long)f2bf(a4.w) << 48);
    crow[32 + lane] = pk;
  }
  __syncthreads();

  int n  = lane & 15;
  int mb = (lane >> 4) << 3;

#pragma unroll
  for (int nt = 0; nt < 8; ++nt) {
    floatx8 acc;
    float bias = b1[nt * 16 + n];
#pragma unroll
    for (int v = 0; v < 8; ++v) acc[v] = bias;
#pragma unroll
    for (int kc = 0; kc < 8; ++kc) {
      FragB16 a = load_a_frag(cat, 256, kc * 32, lane);
      FragB16 b = load_b_frag(pw1, kc * 8 + nt, lane);
      acc = wmma_bf16(a, b, acc);
    }
#pragma unroll
    for (int v = 0; v < 8; ++v)
      hid[(mb + v) * 128 + nt * 16 + n] = f2bf(silu_f(acc[v]));
  }
  __syncthreads();

#pragma unroll
  for (int nt = 0; nt < 8; ++nt) {
    floatx8 acc;
    float bias = b2[nt * 16 + n];
#pragma unroll
    for (int v = 0; v < 8; ++v) acc[v] = bias;
#pragma unroll
    for (int kc = 0; kc < 4; ++kc) {
      FragB16 a = load_a_frag(hid, 128, kc * 32, lane);
      FragB16 b = load_b_frag(pw2, kc * 8 + nt, lane);
      acc = wmma_bf16(a, b, acc);
    }
#pragma unroll
    for (int v = 0; v < 8; ++v) {
      int m = mb + v;
      outf[m * 128 + nt * 16 + n] =
          acc[v] + bf2f(cat[m * 256 + nt * 16 + n]);   // residual h
    }
  }
  __syncthreads();

  if (lane < 16) {
    const float* row = outf + lane * 128;
    float s = 0.0f, ss = 0.0f;
    for (int c = 0; c < 128; ++c) { float x = row[c]; s += x; ss += x * x; }
    float mean = s * (1.0f / 128.0f);
    float var  = ss * (1.0f / 128.0f) - mean * mean;
    stat[lane * 2]     = mean;
    stat[lane * 2 + 1] = rsqrtf(var + 1e-5f);
  }
  __syncthreads();

  for (int idx = lane; idx < 16 * 128; idx += 32) {
    int r = idx >> 7, c = idx & 127;
    long nid = base + r;
    if (nid < N) {
      float y = (outf[idx] - stat[r * 2]) * stat[r * 2 + 1] * gam[c] + bet[c];
      h[(size_t)nid * 128 + c] = f2bf(y);
    }
  }
}

// ---------------------------------------------------------------------------
// Decoders: flow (128->4->4), heat (128->1->1), spec (128->1->1). Tiny VALU.
// ---------------------------------------------------------------------------
__global__ void __launch_bounds__(256)
decoder_kernel(const unsigned short* __restrict__ h,
               const float* __restrict__ fw1, const float* __restrict__ fb1,
               const float* __restrict__ fw2, const float* __restrict__ fb2,
               const float* __restrict__ hw1, const float* __restrict__ hb1,
               const float* __restrict__ hw2, const float* __restrict__ hb2,
               const float* __restrict__ sw1, const float* __restrict__ sb1,
               const float* __restrict__ sw2, const float* __restrict__ sb2,
               float* __restrict__ out, int N) {
  int n = blockIdx.x * blockDim.x + threadIdx.x;
  if (n >= N) return;
  const unsigned short* hr = h + (size_t)n * 128;
  float a0 = fb1[0], a1 = fb1[1], a2 = fb1[2], a3 = fb1[3];
  float ah = hb1[0], as = sb1[0];
  for (int c = 0; c < 128; ++c) {
    float x = bf2f(hr[c]);
    a0 += x * fw1[c * 4 + 0];
    a1 += x * fw1[c * 4 + 1];
    a2 += x * fw1[c * 4 + 2];
    a3 += x * fw1[c * 4 + 3];
    ah += x * hw1[c];
    as += x * sw1[c];
  }
  float t0 = silu_f(a0), t1 = silu_f(a1), t2 = silu_f(a2), t3 = silu_f(a3);
#pragma unroll
  for (int j = 0; j < 4; ++j)
    out[(size_t)n * 4 + j] =
        t0 * fw2[0 * 4 + j] + t1 * fw2[1 * 4 + j] +
        t2 * fw2[2 * 4 + j] + t3 * fw2[3 * 4 + j] + fb2[j];
  out[(size_t)N * 4 + n] = silu_f(ah) * hw2[0] + hb2[0];
  out[(size_t)N * 5 + n] = silu_f(as) * sw2[0] + sb2[0];
}

// ---------------------------------------------------------------------------
extern "C" void kernel_launch(void* const* d_in, const int* in_sizes, int n_in,
                              void* d_out, int out_size, void* d_ws, size_t ws_size,
                              hipStream_t stream) {
  const int N = in_sizes[0] / 16;   // x: [N,16]
  const int E = in_sizes[1] / 2;    // edge_index: [2,E]
  const int L = 8;

  const float* x   = (const float*)d_in[0];
  const int*   ei  = (const int*)d_in[1];
  const float* ea  = (const float*)d_in[2];
  const float* nw1 = (const float*)d_in[3];  const float* nb1 = (const float*)d_in[4];
  const float* nw2 = (const float*)d_in[5];  const float* nb2 = (const float*)d_in[6];
  const float* ew1 = (const float*)d_in[7];  const float* eb1 = (const float*)d_in[8];
  const float* ew2 = (const float*)d_in[9];  const float* eb2 = (const float*)d_in[10];
  const float* pew1 = (const float*)d_in[11]; const float* peb1 = (const float*)d_in[12];
  const float* pew2 = (const float*)d_in[13]; const float* peb2 = (const float*)d_in[14];
  const float* pnw1 = (const float*)d_in[15]; const float* pnb1 = (const float*)d_in[16];
  const float* pnw2 = (const float*)d_in[17]; const float* pnb2 = (const float*)d_in[18];
  const float* eng = (const float*)d_in[19]; const float* enb = (const float*)d_in[20];
  const float* nng = (const float*)d_in[21]; const float* nnb = (const float*)d_in[22];
  const float* fw1 = (const float*)d_in[23]; const float* fb1 = (const float*)d_in[24];
  const float* fw2 = (const float*)d_in[25]; const float* fb2 = (const float*)d_in[26];
  const float* hw1 = (const float*)d_in[27]; const float* hb1 = (const float*)d_in[28];
  const float* hw2 = (const float*)d_in[29]; const float* hb2 = (const float*)d_in[30];
  const float* sw1 = (const float*)d_in[31]; const float* sb1 = (const float*)d_in[32];
  const float* sw2 = (const float*)d_in[33]; const float* sb2 = (const float*)d_in[34];

  const int* srcI = ei;
  const int* dstI = ei + E;

  // workspace partition (bytes)
  char* ws = (char*)d_ws;
  size_t off = 0;
  auto take = [&](size_t bytes) { char* p = ws + off; off = (off + bytes + 255) & ~(size_t)255; return p; };
  unsigned short* h   = (unsigned short*)take((size_t)N * 128 * 2);
  unsigned short* e   = (unsigned short*)take((size_t)E * 128 * 2);
  float*          agg = (float*)take((size_t)N * 128 * 4);
  unsigned int* pwn1  = (unsigned int*)take(2048  * 4);
  unsigned int* pwn2  = (unsigned int*)take(8192  * 4);
  unsigned int* pwe1  = (unsigned int*)take(2048  * 4);
  unsigned int* pwe2  = (unsigned int*)take(8192  * 4);
  unsigned int* ppe1  = (unsigned int*)take((size_t)L * 24576 * 4);
  unsigned int* ppe2  = (unsigned int*)take((size_t)L * 8192  * 4);
  unsigned int* ppn1  = (unsigned int*)take((size_t)L * 16384 * 4);
  unsigned int* ppn2  = (unsigned int*)take((size_t)L * 8192  * 4);
  (void)ws_size; (void)n_in; (void)out_size;

  auto pack = [&](const float* W, unsigned int* dst, int Ksrc, int Kpad) {
    int tot = (Kpad / 32) * 8 * 256;
    pack_w_kernel<<<(tot + 255) / 256, 256, 0, stream>>>(W, dst, Ksrc, Kpad);
  };
  pack(nw1, pwn1, 16, 32);
  pack(nw2, pwn2, 128, 128);
  pack(ew1, pwe1, 4, 32);
  pack(ew2, pwe2, 128, 128);
  for (int l = 0; l < L; ++l) {
    pack(pew1 + (size_t)l * 384 * 128, ppe1 + (size_t)l * 24576, 384, 384);
    pack(pew2 + (size_t)l * 128 * 128, ppe2 + (size_t)l * 8192, 128, 128);
    pack(pnw1 + (size_t)l * 256 * 128, ppn1 + (size_t)l * 16384, 256, 256);
    pack(pnw2 + (size_t)l * 128 * 128, ppn2 + (size_t)l * 8192, 128, 128);
  }

  int tilesN = (N + 15) / 16, tilesE = (E + 15) / 16;
  encoder_kernel<<<(tilesN + 7) / 8, 256, 0, stream>>>(x, 16, N, pwn1, nb1, pwn2, nb2, h);
  encoder_kernel<<<(tilesE + 7) / 8, 256, 0, stream>>>(ea, 4, E, pwe1, eb1, pwe2, eb2, e);

  size_t smem_edge = 4 * (16 * 384 * 2 + 16 * 128 * 2 + 16 * 128 * 4 + 32 * 4 + 16 * 4);
  size_t smem_node = 4 * (16 * 256 * 2 + 16 * 128 * 2 + 16 * 128 * 4 + 32 * 4);

  for (int l = 0; l < L; ++l) {
    zero_kernel<<<1024, 256, 0, stream>>>(agg, (long)N * 128);
    edge_update_kernel<<<(tilesE + 3) / 4, 128, smem_edge, stream>>>(
        h, e, srcI, dstI,
        ppe1 + (size_t)l * 24576, peb1 + (size_t)l * 128,
        ppe2 + (size_t)l * 8192,  peb2 + (size_t)l * 128,
        eng + (size_t)l * 128, enb + (size_t)l * 128, agg, E);
    node_update_kernel<<<(tilesN + 3) / 4, 128, smem_node, stream>>>(
        h, agg,
        ppn1 + (size_t)l * 16384, pnb1 + (size_t)l * 128,
        ppn2 + (size_t)l * 8192,  pnb2 + (size_t)l * 128,
        nng + (size_t)l * 128, nnb + (size_t)l * 128, N);
  }

  decoder_kernel<<<(N + 255) / 256, 256, 0, stream>>>(
      h, fw1, fb1, fw2, fb2, hw1, hb1, hw2, hb2, sw1, sb1, sw2, sb2,
      (float*)d_out, N);
}
